// RadialBasis_51316269253437
// MI455X (gfx1250) — matrix operand order repaired and checked by reading, not compiled, not run on hardware
//
#include <hip/hip_runtime.h>

// MI455X / gfx1250, wave32, bf16 WMMA 16x16x32.
//
// Roofline: ref = 38.4 GFLOP, HBM traffic 0.87 GB -> 37us floor @ 23.3 TB/s.
// All-species dense trick (x4) + tile padding -> ~184 GFLOP of WMMA work,
// ~74us at ~2.5 PF dense bf16: compute & memory balanced, so minimize
// non-WMMA overhead. bf16 weight fragments cached in LDS (built once per
// block per l, fetched as ds_load_b128 pairs), fast-rcp SiLU, stride-32
// transpose tiles (lets the backend merge stores into ds_store_2addr_b32),
// and wave-local ordering: WAR edges use a pure scheduling barrier (LDS is
// serviced in order per wave), RAW cross-lane edges use s_wait_dscnt 0.

#define NPAIRS   1500000
#define NTILES   ((NPAIRS + 15) / 16)   // 93750
#define HIDDEN   32
#define NSPEC    4
#define WAVES    8
#define TOTCOLS  72
#define NBLOCKS  1024
#define TSTRIDE  32                     // LDS tile row stride (dwords)

typedef __attribute__((ext_vector_type(16))) __bf16 v16bf;
typedef __attribute__((ext_vector_type(8)))  float  v8f;

struct Weights { const float* w[16]; };   // w[4*l + j], j: 0=w1,1=w2,2=w3,3=w4

// RAW (cross-lane) edge: drain DS so other lanes' stores are complete.
__device__ __forceinline__ void wave_lds_fence() {
    asm volatile("s_wait_dscnt 0x0" ::: "memory");
}
// WAR edge: LDS is serviced in order per wave; only stop compiler reordering.
__device__ __forceinline__ void wave_sched_barrier() {
    asm volatile("" ::: "memory");
}

__device__ __forceinline__ float silu_fast(float x) {
    // x * sigmoid(x) via v_rcp_f32 (no IEEE division macro)
    return x * __builtin_amdgcn_rcpf(1.0f + __expf(-x));
}

// SiLU the two f32 C fragments (cols 0-15 / 16-31), relayout through per-wave
// LDS scratch into a bf16 A fragment (CDNA5 16-bit A layout). Wave-local only.
__device__ __forceinline__ v16bf silu_transpose(v8f c0, v8f c1,
                                                float* scr, int nc, int half) {
    wave_sched_barrier();   // prior A-reads of scr are ordered before stores
#pragma unroll
    for (int r = 0; r < 8; ++r) {
        const int m = r + half * 8;            // C: lanes16-31 hold M=8..15
        scr[m * TSTRIDE + nc]      = silu_fast(c0[r]);
        scr[m * TSTRIDE + 16 + nc] = silu_fast(c1[r]);
    }
    wave_lds_fence();       // cross-lane RAW: stores visible before reads
    v16bf a;
#pragma unroll
    for (int e = 0; e < 16; ++e) {
        const int v = e >> 1, j = e & 1;
        const int k = 2 * v + ((v >= 4) ? 8 : 0) + half * 8 + j;
        a[e] = (__bf16)scr[nc * TSTRIDE + k];  // A row M = lane&15
    }
    return a;
}

__global__ void __launch_bounds__(32 * WAVES)
rbmlp_kernel(const float* __restrict__ radial,
             const int*   __restrict__ species,
             Weights W,
             float* __restrict__ out)
{
    // Fragment cache: 32 frags (4 species x 4 layers x 2 N-tiles) per l.
    // Element order: [fid][lane][e] -> contiguous 32B per (fid,lane).
    __shared__ __align__(32) __bf16 wlds[32 * 512];
    __shared__ float scratch[WAVES][16 * TSTRIDE];
    __shared__ float otile[WAVES][16 * TSTRIDE];
    __shared__ int   sp[WAVES][16];

    const int lane = threadIdx.x & 31;
    const int wave = threadIdx.x >> 5;
    const int nc   = lane & 15;
    const int half = lane >> 4;

    const int NL[4]   = {24, 20, 16, 12};
    const int COFF[4] = {0, 24, 44, 60};

#pragma unroll
    for (int l = 0; l < 4; ++l) {
        const int n    = NL[l];
        const int coff = COFF[l];
        const float* w1 = W.w[4 * l + 0];
        const float* w2 = W.w[4 * l + 1];
        const float* w3 = W.w[4 * l + 2];
        const float* w4 = W.w[4 * l + 3];

        // ---- build bf16 fragment cache for this l (block-cooperative)
        __syncthreads();
        for (int idx = threadIdx.x; idx < 32 * 512; idx += 32 * WAVES) {
            const int fid   = idx >> 9;
            const int ln    = (idx >> 4) & 31;
            const int e     = idx & 15;
            const int s     = fid >> 3;
            const int layer = (fid >> 1) & 3;
            const int t     = fid & 1;
            const int N = t * 16 + (ln & 15);      // B: lane%16 = column
            const int K = (ln >> 4) * 8 + (e & 7) + (e >> 3) * 16;      // B: lane half splits K
            float v;
            if (layer == 0)      v = (K < n) ? w1[(s * n + K) * HIDDEN + N] : 0.0f;
            else if (layer == 1) v = w2[(s * HIDDEN + K) * HIDDEN + N];
            else if (layer == 2) v = w3[(s * HIDDEN + K) * HIDDEN + N];
            else                 v = (N < n) ? w4[(s * HIDDEN + K) * n + N] : 0.0f;
            wlds[idx] = (__bf16)v;
        }
        __syncthreads();

        const v16bf* wf = (const v16bf*)wlds;   // wf[fid*32 + lane]
        float* scr = &scratch[wave][0];

        // ---- grid-strided per-wave tile loop (no block barriers inside)
        for (int tile = blockIdx.x * WAVES + wave; tile < NTILES;
             tile += gridDim.x * WAVES) {
            const int pair_base = tile * 16;

            if (lane < 16) {
                int p = pair_base + lane;
                if (p >= NPAIRS) p = NPAIRS - 1;
                sp[wave][lane] = species[p];
            }

            // stage x tile (branchless, clamped, zero-padded to K=32)
            wave_sched_barrier();
#pragma unroll
            for (int it = 0; it < 16; ++it) {
                const int idx = it * 32 + lane;
                const int m = idx >> 5, c = idx & 31;
                int p = pair_base + m; if (p >= NPAIRS) p = NPAIRS - 1;
                const int cc = (c < n) ? c : 0;
                const float v = radial[p * TOTCOLS + coff + cc];
                scr[m * TSTRIDE + c] = (c < n) ? v : 0.0f;
            }
            wave_lds_fence();

            // A0 fragment, kept in regs across all species
            v16bf a0;
#pragma unroll
            for (int e = 0; e < 16; ++e) {
                const int v = e >> 1, j = e & 1;
                const int k = 2 * v + ((v >= 4) ? 8 : 0) + half * 8 + j;
                a0[e] = (__bf16)scr[nc * TSTRIDE + k];
            }

            for (int s = 0; s < NSPEC; ++s) {
                const int fb = s * 8 * 32 + lane;

                // layer 1
                v16bf b0 = wf[fb + 0 * 32];
                v16bf b1 = wf[fb + 1 * 32];
                v8f c0 = {}, c1 = {};
                c0 = __builtin_amdgcn_wmma_f32_16x16x32_bf16(false, a0, false, b0,
                                                             (short)0, c0, false, false);
                c1 = __builtin_amdgcn_wmma_f32_16x16x32_bf16(false, a0, false, b1,
                                                             (short)0, c1, false, false);
                v16bf a = silu_transpose(c0, c1, scr, nc, half);

                // layer 2
                b0 = wf[fb + 2 * 32];
                b1 = wf[fb + 3 * 32];
                c0 = (v8f){}; c1 = (v8f){};
                c0 = __builtin_amdgcn_wmma_f32_16x16x32_bf16(false, a, false, b0,
                                                             (short)0, c0, false, false);
                c1 = __builtin_amdgcn_wmma_f32_16x16x32_bf16(false, a, false, b1,
                                                             (short)0, c1, false, false);
                a = silu_transpose(c0, c1, scr, nc, half);

                // layer 3
                b0 = wf[fb + 4 * 32];
                b1 = wf[fb + 5 * 32];
                c0 = (v8f){}; c1 = (v8f){};
                c0 = __builtin_amdgcn_wmma_f32_16x16x32_bf16(false, a, false, b0,
                                                             (short)0, c0, false, false);
                c1 = __builtin_amdgcn_wmma_f32_16x16x32_bf16(false, a, false, b1,
                                                             (short)0, c1, false, false);
                a = silu_transpose(c0, c1, scr, nc, half);

                // layer 4 (N masked to n)
                b0 = wf[fb + 6 * 32];
                b1 = wf[fb + 7 * 32];
                c0 = (v8f){}; c1 = (v8f){};
                c0 = __builtin_amdgcn_wmma_f32_16x16x32_bf16(false, a, false, b0,
                                                             (short)0, c0, false, false);
                if (n > 16) {
                    c1 = __builtin_amdgcn_wmma_f32_16x16x32_bf16(false, a, false, b1,
                                                                 (short)0, c1, false, false);
                }

                // species-select into the out tile
#pragma unroll
                for (int r = 0; r < 8; ++r) {
                    const int m = r + half * 8;
                    if (sp[wave][m] == s) {
                        if (nc < n)        otile[wave][m * TSTRIDE + nc]      = c0[r];
                        if ((16 + nc) < n) otile[wave][m * TSTRIDE + 16 + nc] = c1[r];
                    }
                }
            }

            wave_lds_fence();   // cross-lane RAW: otile complete before store
#pragma unroll
            for (int m = 0; m < 16; ++m) {
                const int p = pair_base + m;
                if (lane < n && p < NPAIRS)
                    out[p * TOTCOLS + coff + lane] = otile[wave][m * TSTRIDE + lane];
            }
        }
    }
}

extern "C" void kernel_launch(void* const* d_in, const int* in_sizes, int n_in,
                              void* d_out, int out_size, void* d_ws, size_t ws_size,
                              hipStream_t stream) {
    (void)in_sizes; (void)n_in; (void)out_size; (void)d_ws; (void)ws_size;
    const float* radial = (const float*)d_in[0];
    const int*   spec   = (const int*)d_in[1];
    Weights W;
    for (int i = 0; i < 16; ++i) W.w[i] = (const float*)d_in[2 + i];
    float* out = (float*)d_out;

    hipLaunchKernelGGL(rbmlp_kernel, dim3(NBLOCKS), dim3(32 * WAVES), 0, stream,
                       radial, spec, W, out);
}
